// MOELayer_69458211111677
// MI455X (gfx1250) — matrix operand order repaired and checked by reading, not compiled, not run on hardware
//
#include <hip/hip_runtime.h>
#include <hip/hip_bf16.h>

// ---------------------------------------------------------------------------
// MoE SwiGLU layer for MI455X (gfx1250, wave32, WMMA).
//
// Shapes: x[2,2048,1024] fp32, Wr[1024,8], W1/W3[8,1024,2816], W2[8,2816,1024]
// T=4096 tokens, D=1024, F=2816, E=8, top-2 routing.
//
//  * Weights converted once/launch to bf16 (138 MB -> fits 192 MB L2) and
//    transposed so each WMMA B-fragment is two runs of 8 contiguous bf16 per lane.
//  * MT=32 token tiles: every B-fragment feeds 2 M-tiles -> half the L2
//    weight traffic of an MT=16 scheme.
//  * Experts split 4+4 over gridDim.y into two partial buffers, summed by a
//    deterministic combine kernel (no atomics anywhere).
//  * x tile staged into LDS with gfx1250 async copies
//    (global_load_async_to_lds_b128 + s_wait_asynccnt).
// ---------------------------------------------------------------------------

typedef __bf16 bf16_t;
typedef __attribute__((ext_vector_type(16))) __bf16 v16bf;
typedef __attribute__((ext_vector_type(8)))  __bf16 v8bf;
typedef __attribute__((ext_vector_type(8)))  float  v8f;

#define D_MODEL 1024
#define D_FF    2816
#define NEXP    8
#define NTOK    4096          // 2*2048
#define MT      32            // token-tile (2 WMMA M-tiles)
#define FC      256           // F-chunk staged through LDS (16 waves x 16)

static __device__ __forceinline__ v16bf frag2(const bf16_t* p0, const bf16_t* p1) {
    v8bf lo = *(const v8bf*)p0;
    v8bf hi = *(const v8bf*)p1;
    return __builtin_shufflevector(lo, hi, 0,1,2,3,4,5,6,7,8,9,10,11,12,13,14,15);
}

static __device__ __forceinline__ v8f wmma_bf16(v16bf a, v16bf b, v8f c) {
    return __builtin_amdgcn_wmma_f32_16x16x32_bf16(false, a, false, b, (short)0, c,
                                                   false, false);
}

// ---------------------------------------------------------------------------
// fp32 -> bf16 elementwise convert (for x)
// ---------------------------------------------------------------------------
__global__ void moe_cvt_bf16(const float* __restrict__ in, bf16_t* __restrict__ out,
                             int n) {
    for (int i = blockIdx.x * blockDim.x + threadIdx.x; i < n;
         i += gridDim.x * blockDim.x)
        out[i] = (bf16_t)in[i];
}

// ---------------------------------------------------------------------------
// Tiled transpose + convert: in fp32 [R][C] -> out bf16 [C][R], gridDim.z = E
// ---------------------------------------------------------------------------
__global__ void moe_transpose_cvt(const float* __restrict__ in,
                                  bf16_t* __restrict__ out, int R, int C) {
    __shared__ float tile[32][33];
    const size_t plane = (size_t)R * (size_t)C;
    in  += (size_t)blockIdx.z * plane;
    out += (size_t)blockIdx.z * plane;

    int x = blockIdx.x * 32 + threadIdx.x;
    int y = blockIdx.y * 32 + threadIdx.y;
#pragma unroll
    for (int j = 0; j < 32; j += 8)
        tile[threadIdx.y + j][threadIdx.x] = in[(size_t)(y + j) * C + x];
    __syncthreads();

    int xo = blockIdx.y * 32 + threadIdx.x;
    int yo = blockIdx.x * 32 + threadIdx.y;
#pragma unroll
    for (int j = 0; j < 32; j += 8)
        out[(size_t)(yo + j) * R + xo] = (bf16_t)tile[threadIdx.x][threadIdx.y + j];
}

// ---------------------------------------------------------------------------
// Router: one wave32 per token. comb[t][e] = softmax prob if e in top-2 else 0
// ---------------------------------------------------------------------------
__global__ __launch_bounds__(128) void moe_router(const float* __restrict__ x,
                                                  const float* __restrict__ Wr,
                                                  float* __restrict__ comb) {
    const int t    = blockIdx.x * 4 + (threadIdx.x >> 5);
    const int lane = threadIdx.x & 31;
    const float* xt = x + (size_t)t * D_MODEL;

    float a[NEXP];
#pragma unroll
    for (int e = 0; e < NEXP; ++e) a[e] = 0.0f;

    for (int d = lane; d < D_MODEL; d += 32) {
        float xv = xt[d];
        const float4* wr = (const float4*)(Wr + (size_t)d * NEXP);
        float4 w0 = wr[0], w1 = wr[1];
        a[0] += xv * w0.x; a[1] += xv * w0.y; a[2] += xv * w0.z; a[3] += xv * w0.w;
        a[4] += xv * w1.x; a[5] += xv * w1.y; a[6] += xv * w1.z; a[7] += xv * w1.w;
    }
#pragma unroll
    for (int off = 16; off > 0; off >>= 1)
#pragma unroll
        for (int e = 0; e < NEXP; ++e) a[e] += __shfl_xor(a[e], off, 32);

    if (lane == 0) {
        float mx = a[0];
#pragma unroll
        for (int e = 1; e < NEXP; ++e) mx = fmaxf(mx, a[e]);
        float p[NEXP], s = 0.0f;
#pragma unroll
        for (int e = 0; e < NEXP; ++e) { p[e] = __expf(a[e] - mx); s += p[e]; }
        float inv = 1.0f / s;
        int i1 = 0;
#pragma unroll
        for (int e = 1; e < NEXP; ++e) if (p[e] > p[i1]) i1 = e;
        int i2 = (i1 == 0) ? 1 : 0;
#pragma unroll
        for (int e = 0; e < NEXP; ++e)
            if (e != i1 && p[e] > p[i2]) i2 = e;
#pragma unroll
        for (int e = 0; e < NEXP; ++e)
            comb[(size_t)t * NEXP + e] = (e == i1 || e == i2) ? p[e] * inv : 0.0f;
    }
}

// ---------------------------------------------------------------------------
// Fused SwiGLU FFN. Grid = (T/32, 2), 512 threads (16 waves).
//   blockIdx.y selects 4 experts; partial output -> pbuf[y].
//   wave w owns output cols [w*64, w*64+64) (4 N-tiles) for both M-tiles.
//   Per expert, per 256-wide F chunk:
//     - each wave: 16x16 h,g tiles for both M-tiles (B frag reused 2x)
//     - act = comb * silu(h) * g -> bf16 in LDS
//     - all waves: acc += act @ W2t cols (B frag reused 2x across M)
// ---------------------------------------------------------------------------
__global__ __launch_bounds__(512) void moe_ffn(const bf16_t* __restrict__ xb,
                                               const float*  __restrict__ comb,
                                               const bf16_t* __restrict__ W1t,
                                               const bf16_t* __restrict__ W3t,
                                               const bf16_t* __restrict__ W2t,
                                               float* __restrict__ pbuf) {
    __shared__ bf16_t sX[MT * D_MODEL];   // 64 KB, [m][k]
    __shared__ bf16_t sAct[MT * FC];      // 16 KB, [m][f]
    __shared__ float  sComb[MT];

    const int m0   = blockIdx.x * MT;
    const int tid  = threadIdx.x;
    const int wave = tid >> 5;            // 0..15
    const int lane = tid & 31;
    const int nLoc = lane & 15;
    const int hiH  = lane >> 4;

    // --- async stage of the bf16 x tile into LDS (CDNA5 ASYNCcnt path) ---
    {
        const bf16_t* src = xb + (size_t)m0 * D_MODEL;
        const uint32_t ldsBase = (uint32_t)(uintptr_t)sX;   // low 32b = LDS offset
        const int nchunk = MT * D_MODEL / 8;                // 16B chunks = 4096
        for (int c = tid; c < nchunk; c += 512) {
            uint32_t laddr = ldsBase + (uint32_t)c * 16u;
            uint64_t gaddr = (uint64_t)(uintptr_t)(src + (size_t)c * 8);
            asm volatile("global_load_async_to_lds_b128 %0, %1, off"
                         :: "v"(laddr), "v"(gaddr) : "memory");
        }
        asm volatile("s_wait_asynccnt 0" ::: "memory");
    }
    __syncthreads();

    const v8f vzero = {};
    v8f acc[2][4];
#pragma unroll
    for (int mi = 0; mi < 2; ++mi)
#pragma unroll
        for (int j = 0; j < 4; ++j) acc[mi][j] = vzero;

    // A-fragment bases per M-tile (row = mi*16 + nLoc, k += hiH*8)
    const bf16_t* aX0 = sX + (size_t)nLoc * D_MODEL + hiH * 8;
    const bf16_t* aX1 = sX + (size_t)(16 + nLoc) * D_MODEL + hiH * 8;
    const bf16_t* aA0 = sAct + (size_t)nLoc * FC + hiH * 8;
    const bf16_t* aA1 = sAct + (size_t)(16 + nLoc) * FC + hiH * 8;

    const int eBase = blockIdx.y * 4;
    for (int ei = 0; ei < 4; ++ei) {
        const int e = eBase + ei;
        if (tid < MT) sComb[tid] = comb[(size_t)(m0 + tid) * NEXP + e];
        __syncthreads();

        const bf16_t* w1e = W1t + (size_t)e * D_FF * D_MODEL;   // [F][D]
        const bf16_t* w3e = W3t + (size_t)e * D_FF * D_MODEL;   // [F][D]
        const bf16_t* w2e = W2t + (size_t)e * D_MODEL * D_FF;   // [D][F]

        for (int fc = 0; fc < D_FF; fc += FC) {
            // -------- gate / up projection --------
            const int f0 = fc + wave * 16;
            const bf16_t* b1p = w1e + (size_t)(f0 + nLoc) * D_MODEL + hiH * 8;
            const bf16_t* b3p = w3e + (size_t)(f0 + nLoc) * D_MODEL + hiH * 8;

            v8f h0 = vzero, h1 = vzero, g0 = vzero, g1 = vzero;
#pragma unroll 4
            for (int k = 0; k < D_MODEL; k += 32) {
                v16bf aF0 = frag2(aX0 + k, aX0 + k + 16);
                v16bf aF1 = frag2(aX1 + k, aX1 + k + 16);
                v16bf b1  = frag2(b1p + k, b1p + k + 16);
                h0 = wmma_bf16(aF0, b1, h0);
                h1 = wmma_bf16(aF1, b1, h1);
                v16bf b3  = frag2(b3p + k, b3p + k + 16);
                g0 = wmma_bf16(aF0, b3, g0);
                g1 = wmma_bf16(aF1, b3, g1);
            }

            // act = comb * silu(h) * g  (C layout: VGPR r -> M=r or r+8)
#pragma unroll
            for (int r = 0; r < 8; ++r) {
                int mrow = r + 8 * hiH;
                float h = h0[r], g = g0[r];
                float s = h / (1.0f + __expf(-h));
                sAct[mrow * FC + (wave * 16 + nLoc)] =
                    (bf16_t)(s * g * sComb[mrow]);
                h = h1[r]; g = g1[r];
                s = h / (1.0f + __expf(-h));
                sAct[(16 + mrow) * FC + (wave * 16 + nLoc)] =
                    (bf16_t)(s * g * sComb[16 + mrow]);
            }
            __syncthreads();

            // -------- down projection --------
#pragma unroll
            for (int j = 0; j < 4; ++j) {
                const int n0 = wave * 64 + j * 16;
                const bf16_t* b2p =
                    w2e + (size_t)(n0 + nLoc) * D_FF + fc + hiH * 8;
#pragma unroll 4
                for (int kk = 0; kk < FC; kk += 32) {
                    v16bf b2  = frag2(b2p + kk, b2p + kk + 16);
                    v16bf aF0 = frag2(aA0 + kk, aA0 + kk + 16);
                    acc[0][j] = wmma_bf16(aF0, b2, acc[0][j]);
                    v16bf aF1 = frag2(aA1 + kk, aA1 + kk + 16);
                    acc[1][j] = wmma_bf16(aF1, b2, acc[1][j]);
                }
            }
            __syncthreads();   // sAct reuse / sComb rewrite
        }
    }

    // Partial output for this expert group (single writer, no atomics).
    float* pout = pbuf + (size_t)blockIdx.y * NTOK * D_MODEL;
#pragma unroll
    for (int mi = 0; mi < 2; ++mi)
#pragma unroll
        for (int j = 0; j < 4; ++j) {
            int n = wave * 64 + j * 16 + nLoc;
#pragma unroll
            for (int r = 0; r < 8; ++r) {
                int t = m0 + mi * 16 + r + 8 * hiH;
                pout[(size_t)t * D_MODEL + n] = acc[mi][j][r];
            }
        }
}

// ---------------------------------------------------------------------------
// Deterministic combine of the two expert-group partials.
// ---------------------------------------------------------------------------
__global__ void moe_combine(const float* __restrict__ p0,
                            const float* __restrict__ p1,
                            float* __restrict__ out, int nvec) {
    const float4* a = (const float4*)p0;
    const float4* b = (const float4*)p1;
    float4* o = (float4*)out;
    for (int i = blockIdx.x * blockDim.x + threadIdx.x; i < nvec;
         i += gridDim.x * blockDim.x) {
        float4 u = a[i], v = b[i];
        o[i] = make_float4(u.x + v.x, u.y + v.y, u.z + v.z, u.w + v.w);
    }
}

// ---------------------------------------------------------------------------
// Host launcher
// ---------------------------------------------------------------------------
extern "C" void kernel_launch(void* const* d_in, const int* in_sizes, int n_in,
                              void* d_out, int out_size, void* d_ws, size_t ws_size,
                              hipStream_t stream) {
    const float* x  = (const float*)d_in[0];   // [T, D]
    const float* Wr = (const float*)d_in[1];   // [D, 8]
    const float* W1 = (const float*)d_in[2];   // [E, D, F]
    const float* W3 = (const float*)d_in[3];   // [E, D, F]
    const float* W2 = (const float*)d_in[4];   // [E, F, D]
    float* out = (float*)d_out;

    char* ws = (char*)d_ws;
    size_t off = 0;
    auto carve = [&](size_t bytes) -> void* {
        void* p = ws + off;
        off += (bytes + 255) & ~(size_t)255;
        return p;
    };
    bf16_t* xb   = (bf16_t*)carve((size_t)NTOK * D_MODEL * sizeof(bf16_t));
    float*  comb = (float*) carve((size_t)NTOK * NEXP * sizeof(float));
    const size_t wbytes = (size_t)NEXP * D_MODEL * D_FF * sizeof(bf16_t);
    bf16_t* W1t = (bf16_t*)carve(wbytes);   // [E][F][D]
    bf16_t* W3t = (bf16_t*)carve(wbytes);   // [E][F][D]
    bf16_t* W2t = (bf16_t*)carve(wbytes);   // [E][D][F]
    float*  pbuf = (float*)carve((size_t)2 * NTOK * D_MODEL * sizeof(float));

    // 1) x -> bf16
    moe_cvt_bf16<<<1024, 256, 0, stream>>>(x, xb, NTOK * D_MODEL);

    // 2) router -> dense combine weights
    moe_router<<<NTOK / 4, 128, 0, stream>>>(x, Wr, comb);

    // 3) weight transpose+convert (bf16 weights then live in L2)
    dim3 tb(32, 8, 1);
    moe_transpose_cvt<<<dim3(D_FF / 32, D_MODEL / 32, NEXP), tb, 0, stream>>>(
        W1, W1t, D_MODEL, D_FF);
    moe_transpose_cvt<<<dim3(D_FF / 32, D_MODEL / 32, NEXP), tb, 0, stream>>>(
        W3, W3t, D_MODEL, D_FF);
    moe_transpose_cvt<<<dim3(D_MODEL / 32, D_FF / 32, NEXP), tb, 0, stream>>>(
        W2, W2t, D_FF, D_MODEL);

    // 4) fused WMMA SwiGLU MoE: experts 0-3 -> pbuf[0], experts 4-7 -> pbuf[1]
    moe_ffn<<<dim3(NTOK / MT, 2), 512, 0, stream>>>(xb, comb, W1t, W3t, W2t, pbuf);

    // 5) deterministic sum of the two partials
    const size_t half = (size_t)NTOK * D_MODEL;
    moe_combine<<<2048, 256, 0, stream>>>(pbuf, pbuf + half, out,
                                          (int)(half / 4));
}
